// EquivariantGridModel_30674656428247
// MI455X (gfx1250) — compile-verified, matrix operand order checked by reading, not run on hardware
//
#include <hip/hip_runtime.h>
#include <hip/hip_bf16.h>
#include <math.h>

typedef __attribute__((ext_vector_type(16))) _Float16 v16h;
typedef __attribute__((ext_vector_type(8)))  _Float16 v8h;
typedef __attribute__((ext_vector_type(8)))  float    v8f;

#define DEVFN static __device__ __forceinline__

DEVFN float silu_f(float v) { return v / (1.0f + __expf(-v)); }

DEVFN v8f wmma_f16(v16h a, v16h b, v8f c) {
  return __builtin_amdgcn_wmma_f32_16x16x32_f16(false, a, false, b, (short)0, c,
                                                false, false);
}

DEVFN v16h pack2(v8h lo, v8h hi) {
  union { v16h v; v8h h[2]; } u;
  u.h[0] = lo; u.h[1] = hi;
  return u.v;
}

// Per-lane A fragment for one 32-wide K chunk, loaded directly from a row-major
// f16 row pointer: lane (m, half) needs K = {b0..b0+7, b0+16..b0+23} -> two
// contiguous 16B runs.
DEVFN v16h row_afrag(const _Float16* rowp, int kbase, int b0) {
  const _Float16* p = rowp + kbase + b0;
  return pack2(*(const v8h*)p, *(const v8h*)(p + 16));
}

// Fragment-major packed B: frag(c,t) for lane l is 16 contiguous f16 at
// Wf + ((c*8 + t)*32 + l)*16.
DEVFN v16h packed_bfrag(const _Float16* Wf, int c, int nt, int lane) {
  return *(const v16h*)(Wf + (size_t)(((c * 8 + nt) * 32 + lane) * 16));
}

// Position of element (row m, feature k) inside a fragment-layout LDS buffer
// (chunks of 32 K; 32 lanes x 16 f16 per chunk).
DEVFN int frag_idx(int m, int k) {
  int c = k >> 5, kk = k & 31;
  int lanehalf = (kk >> 3) & 1;
  int elem = ((kk >> 4) & 1) * 8 + (kk & 7);
  return c * 512 + (m + lanehalf * 16) * 16 + elem;
}

// ---------------- prep kernels ----------------

// Pack row-major [krows(pad kpad)][128] f32 weights into fragment-major f16:
// dst[((c*8 + t)*32 + l)*16 + e] = W[k][t*16 + (l&15)],
// k = c*32 + ((l&16)?8:0) + e + (e>=8 ? 8 : 0)
__global__ void k_cvt_bfrag(const float* __restrict__ src, _Float16* __restrict__ dst,
                            int krows, int kpad) {
  int i = blockIdx.x * blockDim.x + threadIdx.x;
  if (i >= kpad * 128) return;
  int e = i & 15;
  int l = (i >> 4) & 31;
  int t = (i >> 9) & 7;
  int c = i >> 12;
  int n = t * 16 + (l & 15);
  int b0 = (l & 16) ? 8 : 0;
  int k = c * 32 + b0 + e + (e >= 8 ? 8 : 0);
  dst[i] = (k < krows) ? (_Float16)src[k * 128 + n] : (_Float16)0.0f;
}

__global__ void k_embed(const int* __restrict__ atoms, const float* __restrict__ emb,
                        _Float16* __restrict__ hh, int n) {
  int i = blockIdx.x * blockDim.x + threadIdx.x;
  if (i >= n * 128) return;
  int node = i >> 7, c = i & 127;
  hh[i] = (_Float16)emb[atoms[node] * 128 + c];
}

// ---------------- EquivariantLayer: fused dist + 2x MLP(257->128->1) + scatter ----------------

__global__ __launch_bounds__(32) void k_eq(
    const _Float16* __restrict__ hh, const float* __restrict__ x,
    const int* __restrict__ ei, int ne,
    const _Float16* W1a, const float* b1a, const float* W2a, const float* b2a,
    const _Float16* W1b, const float* b1b, const float* W2b, const float* b2b,
    float* __restrict__ v1, float* __restrict__ v2) {
  __shared__ _Float16 thid[16 * 128];  // hidden tile, fragment layout

  int lane = threadIdx.x;
  int e0 = blockIdx.x * 16;
  int m = lane & 15;
  int b0 = (lane & 16) ? 8 : 0;

  int e = e0 + m;
  bool valid = (e < ne);
  if (!valid) e = ne - 1;
  int r = ei[e], cD = ei[ne + e];
  float dx = x[r * 3 + 0] - x[cD * 3 + 0];
  float dy = x[r * 3 + 1] - x[cD * 3 + 1];
  float dz = x[r * 3 + 2] - x[cD * 3 + 2];
  float dist = sqrtf(dx * dx + dy * dy + dz * dz);

  // A fragments for x_ij = [h[row](128) | h[col](128) | dist | pad..287]
  const _Float16* hr = hh + (size_t)r * 128;
  const _Float16* hc = hh + (size_t)cD * 128;
  v16h A[9];
#pragma unroll
  for (int c = 0; c < 4; ++c) A[c] = row_afrag(hr, c * 32, b0);
#pragma unroll
  for (int c = 0; c < 4; ++c) A[4 + c] = row_afrag(hc, c * 32, b0);
  {
    v16h a8 = {};
    if (lane < 16) a8[0] = (_Float16)dist;  // k=256 lives in elem 0 of lanes 0-15
    A[8] = a8;
  }

  float mes0 = 0.f, mes1 = 0.f;
  for (int p = 0; p < 2; ++p) {
    const _Float16* W1 = p ? W1b : W1a;
    const float* bb1 = p ? b1b : b1a;
    const float* W2 = p ? W2b : W2a;
    float bb2 = p ? b2b[0] : b2a[0];

    for (int nt = 0; nt < 8; ++nt) {
      v8f acc = {};
#pragma unroll
      for (int c = 0; c < 9; ++c)
        acc = wmma_f16(A[c], packed_bfrag(W1, c, nt, lane), acc);
      int n = nt * 16 + (lane & 15);
      float bias = bb1[n];
      int mhi = (lane & 16) ? 8 : 0;
#pragma unroll
      for (int rr = 0; rr < 8; ++rr)
        thid[frag_idx(rr + mhi, n)] = (_Float16)silu_f(acc[rr] + bias);
    }
    __syncthreads();
    if (lane < 16) {
      float s = bb2;
      for (int k = 0; k < 128; ++k) s += (float)thid[frag_idx(lane, k)] * W2[k];
      if (p) mes1 = s; else mes0 = s;
    }
    __syncthreads();
  }

  if (lane < 16 && valid) {
    float rr = dist * (1.0f / 4.5f);
    rr = fminf(fmaxf(rr, 0.0f), 1.0f);
    float r5 = rr * rr * rr * rr * rr;
    float coe = 1.0f - 21.0f * r5 + 35.0f * r5 * rr - 15.0f * r5 * rr * rr;
    coe = fminf(fmaxf(coe, 0.0f), 1.0f);
    float inv = 1.0f / (dist + 1e-8f);
    float f1 = coe * mes0 * inv, f2 = coe * mes1 * inv;
    atomicAdd(&v1[cD * 3 + 0], dx * f1);
    atomicAdd(&v1[cD * 3 + 1], dy * f1);
    atomicAdd(&v1[cD * 3 + 2], dz * f1);
    atomicAdd(&v2[cD * 3 + 0], dx * f2);
    atomicAdd(&v2[cD * 3 + 1], dy * f2);
    atomicAdd(&v2[cD * 3 + 2], dz * f2);
  }
}

// ---------------- Gram-Schmidt orientation matrices ----------------

__global__ void k_gram(const float* __restrict__ v1, const float* __restrict__ v2,
                       float* __restrict__ R, int n) {
  int i = blockIdx.x * blockDim.x + threadIdx.x;
  if (i >= n) return;
  const float EPSF = 1e-6f;
  float a0 = v1[i * 3], a1 = v1[i * 3 + 1], a2 = v1[i * 3 + 2];
  float b0 = v2[i * 3], b1 = v2[i * 3 + 1], b2 = v2[i * 3 + 2];
  float an = sqrtf(a0 * a0 + a1 * a1 + a2 * a2);
  float n10 = a0 / (an + EPSF), n11 = a1 / (an + EPSF), n12 = a2 / (an + EPSF);
  float d = n10 * b0 + n11 * b1 + n12 * b2;
  float p0 = b0 - d * n10, p1 = b1 - d * n11, p2 = b2 - d * n12;
  float pn = sqrtf(p0 * p0 + p1 * p1 + p2 * p2);
  float dd0 = -n11, dd1 = n10;
  float dn = sqrtf(dd0 * dd0 + dd1 * dd1);
  float e0, e1, e2;
  if (dn > EPSF) { e0 = dd0 / (dn + EPSF); e1 = dd1 / (dn + EPSF); e2 = 0.f; }
  else           { e0 = 0.f; e1 = 0.f; e2 = 1.f; }
  float n20, n21, n22;
  if (pn > EPSF) { n20 = p0 / (pn + EPSF); n21 = p1 / (pn + EPSF); n22 = p2 / (pn + EPSF); }
  else           { n20 = e0; n21 = e1; n22 = e2; }
  float c0 = n11 * n22 - n12 * n21;
  float c1 = n12 * n20 - n10 * n22;
  float c2 = n10 * n21 - n11 * n20;
  float cn = sqrtf(c0 * c0 + c1 * c1 + c2 * c2);
  c0 /= (cn + EPSF); c1 /= (cn + EPSF); c2 /= (cn + EPSF);
  float* Ri = R + (size_t)i * 9;  // R[row][col], cols = (n1, n2, n3)
  Ri[0] = n10; Ri[1] = n20; Ri[2] = c0;
  Ri[3] = n11; Ri[4] = n21; Ri[5] = c1;
  Ri[6] = n12; Ri[7] = n22; Ri[8] = c2;
}

// ---------------- edge attributes for gridification ----------------

__global__ void k_ea(const float* __restrict__ x, const float* __restrict__ gp,
                     const float* __restrict__ R, const int* __restrict__ gsrc,
                     const int* __restrict__ gdst, _Float16* __restrict__ ea, int ne) {
  int e = blockIdx.x * blockDim.x + threadIdx.x;
  if (e >= ne) return;
  int s = gsrc[e], g = gdst[e];
  float px = x[s * 3], py = x[s * 3 + 1], pz = x[s * 3 + 2];
  float rx = gp[g * 3] - px, ry = gp[g * 3 + 1] - py, rz = gp[g * 3 + 2] - pz;
  const float* Rs = R + (size_t)s * 9;
  float t0 = rx * Rs[0] + ry * Rs[3] + rz * Rs[6];
  float t1 = rx * Rs[1] + ry * Rs[4] + rz * Rs[7];
  float t2 = rx * Rs[2] + ry * Rs[5] + rz * Rs[8];
  _Float16* o = ea + (size_t)e * 32;
  o[0] = (_Float16)px; o[1] = (_Float16)py; o[2] = (_Float16)pz;
  o[3] = (_Float16)t0; o[4] = (_Float16)t1; o[5] = (_Float16)t2;
#pragma unroll
  for (int k = 6; k < 32; ++k) o[k] = (_Float16)0.0f;
}

// ---------------- generic fused 2-layer MLP (kin -> 128 -> 128) with WMMA ----------------

template <int NCH>
__global__ __launch_bounds__(32) void k_mlp2(
    const _Float16* __restrict__ X, int nrows,
    const _Float16* __restrict__ W1, const float* __restrict__ b1,
    const _Float16* __restrict__ W2, const float* __restrict__ b2,
    _Float16* __restrict__ out_h, float* __restrict__ out_f) {
  __shared__ _Float16 thid[16 * 128];  // hidden tile, fragment layout
  const int kin = NCH * 32;
  int lane = threadIdx.x;
  int row0 = blockIdx.x * 16;
  int m = lane & 15;
  int b0 = (lane & 16) ? 8 : 0;

  int row = row0 + m;
  if (row >= nrows) row = nrows - 1;
  const _Float16* xr = X + (size_t)row * kin;
  v16h A[NCH];
#pragma unroll
  for (int c = 0; c < NCH; ++c) A[c] = row_afrag(xr, c * 32, b0);

  for (int nt = 0; nt < 8; ++nt) {
    v8f acc = {};
#pragma unroll
    for (int c = 0; c < NCH; ++c)
      acc = wmma_f16(A[c], packed_bfrag(W1, c, nt, lane), acc);
    int n = nt * 16 + (lane & 15);
    float bias = b1[n];
    int mhi = (lane & 16) ? 8 : 0;
#pragma unroll
    for (int rr = 0; rr < 8; ++rr)
      thid[frag_idx(rr + mhi, n)] = (_Float16)silu_f(acc[rr] + bias);
  }
  __syncthreads();

  v16h A2[4];
#pragma unroll
  for (int c = 0; c < 4; ++c)
    A2[c] = *(const v16h*)(thid + (c * 32 + lane) * 16);

  for (int nt = 0; nt < 8; ++nt) {
    v8f acc = {};
#pragma unroll
    for (int c = 0; c < 4; ++c)
      acc = wmma_f16(A2[c], packed_bfrag(W2, c, nt, lane), acc);
    int n = nt * 16 + (lane & 15);
    float bias = b2[n];
    int mhi = (lane & 16) ? 8 : 0;
#pragma unroll
    for (int rr = 0; rr < 8; ++rr) {
      int orow = row0 + rr + mhi;
      if (orow < nrows) {
        float v = acc[rr] + bias;
        if (out_h) out_h[(size_t)orow * 128 + n] = (_Float16)v;
        else       out_f[(size_t)orow * 128 + n] = v;
      }
    }
  }
}

// ---------------- message MLP (gathered 256 -> 128 -> 128) + atomic grid scatter ----------------

__global__ __launch_bounds__(32) void k_msg(
    const _Float16* __restrict__ nf, const _Float16* __restrict__ ef,
    const int* __restrict__ gsrc, const int* __restrict__ gdst, int ne,
    const _Float16* __restrict__ W1, const float* __restrict__ b1,
    const _Float16* __restrict__ W2, const float* __restrict__ b2,
    float* __restrict__ gf) {
  __shared__ _Float16 thid[16 * 128];
  int lane = threadIdx.x;
  int e0 = blockIdx.x * 16;
  int m = lane & 15;
  int b0 = (lane & 16) ? 8 : 0;

  int e = e0 + m;
  if (e >= ne) e = ne - 1;
  const _Float16* nr = nf + (size_t)gsrc[e] * 128;
  const _Float16* er = ef + (size_t)e * 128;
  v16h A[8];
#pragma unroll
  for (int c = 0; c < 4; ++c) A[c] = row_afrag(nr, c * 32, b0);
#pragma unroll
  for (int c = 0; c < 4; ++c) A[4 + c] = row_afrag(er, c * 32, b0);

  for (int nt = 0; nt < 8; ++nt) {
    v8f acc = {};
#pragma unroll
    for (int c = 0; c < 8; ++c)
      acc = wmma_f16(A[c], packed_bfrag(W1, c, nt, lane), acc);
    int n = nt * 16 + (lane & 15);
    float bias = b1[n];
    int mhi = (lane & 16) ? 8 : 0;
#pragma unroll
    for (int rr = 0; rr < 8; ++rr)
      thid[frag_idx(rr + mhi, n)] = (_Float16)silu_f(acc[rr] + bias);
  }
  __syncthreads();

  v16h A2[4];
#pragma unroll
  for (int c = 0; c < 4; ++c)
    A2[c] = *(const v16h*)(thid + (c * 32 + lane) * 16);

  for (int nt = 0; nt < 8; ++nt) {
    v8f acc = {};
#pragma unroll
    for (int c = 0; c < 4; ++c)
      acc = wmma_f16(A2[c], packed_bfrag(W2, c, nt, lane), acc);
    int n = nt * 16 + (lane & 15);
    float bias = b2[n];
    int mhi = (lane & 16) ? 8 : 0;
#pragma unroll
    for (int rr = 0; rr < 8; ++rr) {
      int ee = e0 + rr + mhi;
      if (ee < ne) atomicAdd(&gf[(size_t)gdst[ee] * 128 + n], acc[rr] + bias);
    }
  }
}

__global__ void k_counts(const int* __restrict__ gdst, float* __restrict__ cnt, int ne) {
  int e = blockIdx.x * blockDim.x + threadIdx.x;
  if (e < ne) atomicAdd(&cnt[gdst[e]], 1.0f);
}

__global__ void k_gfdiv(const float* __restrict__ gf, const float* __restrict__ cnt,
                        _Float16* __restrict__ out) {
  int i = blockIdx.x * blockDim.x + threadIdx.x;
  if (i >= 216 * 128) return;
  float c = fmaxf(cnt[i >> 7], 1.0f);
  out[i] = (_Float16)(gf[i] / c);
}

// ---------------- tiny 3D CNN tail ----------------

__global__ void k_conv1(const float* __restrict__ gfo, const float* __restrict__ w,
                        const float* __restrict__ b, float* __restrict__ out) {
  int tid = blockIdx.x * blockDim.x + threadIdx.x;
  if (tid >= 64 * 216) return;
  int co = tid / 216, g = tid % 216;
  int d = g / 36, h = (g / 6) % 6, wv = g % 6;
  float acc = b[co];
  for (int kz = 0; kz < 3; ++kz) {
    int zz = d + kz - 1; if (zz < 0 || zz >= 6) continue;
    for (int ky = 0; ky < 3; ++ky) {
      int yy = h + ky - 1; if (yy < 0 || yy >= 6) continue;
      for (int kx = 0; kx < 3; ++kx) {
        int xx = wv + kx - 1; if (xx < 0 || xx >= 6) continue;
        int gg = zz * 36 + yy * 6 + xx;
        int tap = kz * 9 + ky * 3 + kx;
        const float* ip = gfo + (size_t)gg * 128;
        const float* wp = w + (size_t)co * 128 * 27 + tap;
        float s = 0.f;
        for (int ci = 0; ci < 128; ++ci) s += ip[ci] * wp[(size_t)ci * 27];
        acc += s;
      }
    }
  }
  acc *= rsqrtf(1.0f + 1e-5f);
  out[tid] = fmaxf(acc, 0.0f);
}

__global__ void k_pool1(const float* __restrict__ in, float* __restrict__ out) {
  int tid = blockIdx.x * blockDim.x + threadIdx.x;
  if (tid >= 64 * 27) return;
  int co = tid / 27, p = tid % 27;
  int pd = p / 9, ph = (p / 3) % 3, pw = p % 3;
  float m = -1e30f;
  for (int dz = 0; dz < 2; ++dz)
    for (int dy = 0; dy < 2; ++dy)
      for (int dx = 0; dx < 2; ++dx)
        m = fmaxf(m, in[co * 216 + (2 * pd + dz) * 36 + (2 * ph + dy) * 6 + (2 * pw + dx)]);
  out[tid] = m;
}

__global__ void k_conv2(const float* __restrict__ y1, const float* __restrict__ w,
                        const float* __restrict__ b, float* __restrict__ out) {
  int tid = blockIdx.x * blockDim.x + threadIdx.x;
  if (tid >= 128 * 27) return;
  int co = tid / 27, g = tid % 27;
  int d = g / 9, h = (g / 3) % 3, wv = g % 3;
  float acc = b[co];
  for (int kz = 0; kz < 3; ++kz) {
    int zz = d + kz - 1; if (zz < 0 || zz >= 3) continue;
    for (int ky = 0; ky < 3; ++ky) {
      int yy = h + ky - 1; if (yy < 0 || yy >= 3) continue;
      for (int kx = 0; kx < 3; ++kx) {
        int xx = wv + kx - 1; if (xx < 0 || xx >= 3) continue;
        int gg = zz * 9 + yy * 3 + xx;
        int tap = kz * 9 + ky * 3 + kx;
        for (int ci = 0; ci < 64; ++ci)
          acc += y1[ci * 27 + gg] * w[((size_t)co * 64 + ci) * 27 + tap];
      }
    }
  }
  acc *= rsqrtf(1.0f + 1e-5f);
  out[tid] = fmaxf(acc, 0.0f);
}

__global__ void k_pool2(const float* __restrict__ in, float* __restrict__ out) {
  int co = threadIdx.x;
  if (co >= 128) return;
  float m = -1e30f;
  for (int d = 0; d < 2; ++d)
    for (int h = 0; h < 2; ++h)
      for (int w = 0; w < 2; ++w)
        m = fmaxf(m, in[co * 27 + d * 9 + h * 3 + w]);
  out[co] = m;
}

__global__ void k_conv3(const float* __restrict__ y2, const float* __restrict__ w,
                        const float* __restrict__ b, float* __restrict__ out) {
  int co = threadIdx.x;
  if (co >= 256) return;
  float acc = b[co];
  for (int ci = 0; ci < 128; ++ci)
    acc += y2[ci] * w[((size_t)co * 128 + ci) * 27 + 13];  // center tap only
  acc *= rsqrtf(1.0f + 1e-5f);
  out[co] = fmaxf(acc, 0.0f);
}

__global__ void k_fc(const float* __restrict__ y3, const float* __restrict__ w,
                     const float* __restrict__ b, float* __restrict__ out) {
  int o = threadIdx.x;
  if (o >= 20) return;
  float acc = b[o];
  for (int c = 0; c < 256; ++c) acc += y3[c] * w[c * 20 + o];
  out[o] = acc;
}

// ---------------- host launcher ----------------

extern "C" void kernel_launch(void* const* d_in, const int* in_sizes, int n_in,
                              void* d_out, int out_size, void* d_ws, size_t ws_size,
                              hipStream_t stream) {
  (void)in_sizes; (void)n_in; (void)out_size; (void)ws_size;
  const int N = 20000, E = 320000, EG = 200000, G = 216;

  const int*   atoms = (const int*)d_in[0];
  const float* x     = (const float*)d_in[1];
  const int*   ei    = (const int*)d_in[2];
  const int*   gsrc  = (const int*)d_in[3];
  const int*   gdst  = (const int*)d_in[4];
  const float* gp    = (const float*)d_in[5];
  const float* emb   = (const float*)d_in[6];
  const float* eq1W1 = (const float*)d_in[7];
  const float* eq1b1 = (const float*)d_in[8];
  const float* eq1W2 = (const float*)d_in[9];
  const float* eq1b2 = (const float*)d_in[10];
  const float* eq2W1 = (const float*)d_in[11];
  const float* eq2b1 = (const float*)d_in[12];
  const float* eq2W2 = (const float*)d_in[13];
  const float* eq2b2 = (const float*)d_in[14];
  const float* nodeW1 = (const float*)d_in[15];
  const float* nodeb1 = (const float*)d_in[16];
  const float* nodeW2 = (const float*)d_in[17];
  const float* nodeb2 = (const float*)d_in[18];
  const float* edgeW1 = (const float*)d_in[19];
  const float* edgeb1 = (const float*)d_in[20];
  const float* edgeW2 = (const float*)d_in[21];
  const float* edgeb2 = (const float*)d_in[22];
  const float* msgW1 = (const float*)d_in[23];
  const float* msgb1 = (const float*)d_in[24];
  const float* msgW2 = (const float*)d_in[25];
  const float* msgb2 = (const float*)d_in[26];
  const float* updW1 = (const float*)d_in[27];
  const float* updb1 = (const float*)d_in[28];
  const float* updW2 = (const float*)d_in[29];
  const float* updb2 = (const float*)d_in[30];
  const float* c1w = (const float*)d_in[31];
  const float* c1b = (const float*)d_in[32];
  const float* c2w = (const float*)d_in[33];
  const float* c2b = (const float*)d_in[34];
  const float* c3w = (const float*)d_in[35];
  const float* c3b = (const float*)d_in[36];
  const float* fcw = (const float*)d_in[37];
  const float* fcb = (const float*)d_in[38];
  float* dout = (float*)d_out;

  char* base = (char*)d_ws;
  size_t off = 0;
  auto alloc = [&](size_t bytes) -> char* {
    char* p = base + off;
    off = (off + bytes + 255) & ~(size_t)255;
    return p;
  };
  _Float16* hh      = (_Float16*)alloc((size_t)N * 128 * 2);
  float*    v1      = (float*)alloc((size_t)N * 3 * 4);
  float*    v2      = (float*)alloc((size_t)N * 3 * 4);
  float*    R       = (float*)alloc((size_t)N * 9 * 4);
  _Float16* nfh     = (_Float16*)alloc((size_t)N * 128 * 2);
  _Float16* eah     = (_Float16*)alloc((size_t)EG * 32 * 2);
  _Float16* efh     = (_Float16*)alloc((size_t)EG * 128 * 2);
  float*    gf      = (float*)alloc((size_t)G * 128 * 4);
  float*    cnt     = (float*)alloc((size_t)G * 4);
  _Float16* gfih    = (_Float16*)alloc((size_t)G * 128 * 2);
  float*    gfo     = (float*)alloc((size_t)G * 128 * 4);
  _Float16* eq1W1h  = (_Float16*)alloc((size_t)288 * 128 * 2);
  _Float16* eq2W1h  = (_Float16*)alloc((size_t)288 * 128 * 2);
  _Float16* nodeW1h = (_Float16*)alloc((size_t)128 * 128 * 2);
  _Float16* nodeW2h = (_Float16*)alloc((size_t)128 * 128 * 2);
  _Float16* edgeW1h = (_Float16*)alloc((size_t)32 * 128 * 2);
  _Float16* edgeW2h = (_Float16*)alloc((size_t)128 * 128 * 2);
  _Float16* msgW1h  = (_Float16*)alloc((size_t)256 * 128 * 2);
  _Float16* msgW2h  = (_Float16*)alloc((size_t)128 * 128 * 2);
  _Float16* updW1h  = (_Float16*)alloc((size_t)128 * 128 * 2);
  _Float16* updW2h  = (_Float16*)alloc((size_t)128 * 128 * 2);
  float* act1 = (float*)alloc(64 * 216 * 4);
  float* y1   = (float*)alloc(64 * 27 * 4);
  float* act2 = (float*)alloc(128 * 27 * 4);
  float* y2   = (float*)alloc(128 * 4);
  float* y3   = (float*)alloc(256 * 4);

  auto cdiv = [](int a, int b) { return (a + b - 1) / b; };

  // weight packing (fp32 -> f16, fragment-major, K padded to multiple of 32)
  k_cvt_bfrag<<<cdiv(288 * 128, 256), 256, 0, stream>>>(eq1W1, eq1W1h, 257, 288);
  k_cvt_bfrag<<<cdiv(288 * 128, 256), 256, 0, stream>>>(eq2W1, eq2W1h, 257, 288);
  k_cvt_bfrag<<<cdiv(128 * 128, 256), 256, 0, stream>>>(nodeW1, nodeW1h, 128, 128);
  k_cvt_bfrag<<<cdiv(128 * 128, 256), 256, 0, stream>>>(nodeW2, nodeW2h, 128, 128);
  k_cvt_bfrag<<<cdiv(32 * 128, 256), 256, 0, stream>>>(edgeW1, edgeW1h, 6, 32);
  k_cvt_bfrag<<<cdiv(128 * 128, 256), 256, 0, stream>>>(edgeW2, edgeW2h, 128, 128);
  k_cvt_bfrag<<<cdiv(256 * 128, 256), 256, 0, stream>>>(msgW1, msgW1h, 256, 256);
  k_cvt_bfrag<<<cdiv(128 * 128, 256), 256, 0, stream>>>(msgW2, msgW2h, 128, 128);
  k_cvt_bfrag<<<cdiv(128 * 128, 256), 256, 0, stream>>>(updW1, updW1h, 128, 128);
  k_cvt_bfrag<<<cdiv(128 * 128, 256), 256, 0, stream>>>(updW2, updW2h, 128, 128);

  // node embedding -> f16
  k_embed<<<cdiv(N * 128, 256), 256, 0, stream>>>(atoms, emb, hh, N);

  // zero accumulators
  hipMemsetAsync(v1, 0, (size_t)N * 3 * 4, stream);
  hipMemsetAsync(v2, 0, (size_t)N * 3 * 4, stream);
  hipMemsetAsync(gf, 0, (size_t)G * 128 * 4, stream);
  hipMemsetAsync(cnt, 0, (size_t)G * 4, stream);

  // equivariant layer (big WMMA GEMM + scatter)
  k_eq<<<cdiv(E, 16), 32, 0, stream>>>(hh, x, ei, E,
                                       eq1W1h, eq1b1, eq1W2, eq1b2,
                                       eq2W1h, eq2b1, eq2W2, eq2b2, v1, v2);
  k_gram<<<cdiv(N, 256), 256, 0, stream>>>(v1, v2, R, N);

  // node features
  k_mlp2<4><<<cdiv(N, 16), 32, 0, stream>>>(hh, N, nodeW1h, nodeb1, nodeW2h, nodeb2,
                                            nfh, (float*)nullptr);
  // edge attributes + edge features
  k_ea<<<cdiv(EG, 256), 256, 0, stream>>>(x, gp, R, gsrc, gdst, eah, EG);
  k_mlp2<1><<<cdiv(EG, 16), 32, 0, stream>>>(eah, EG, edgeW1h, edgeb1, edgeW2h, edgeb2,
                                             efh, (float*)nullptr);
  // message passing into grid
  k_counts<<<cdiv(EG, 256), 256, 0, stream>>>(gdst, cnt, EG);
  k_msg<<<cdiv(EG, 16), 32, 0, stream>>>(nfh, efh, gsrc, gdst, EG,
                                         msgW1h, msgb1, msgW2h, msgb2, gf);
  k_gfdiv<<<cdiv(G * 128, 256), 256, 0, stream>>>(gf, cnt, gfih);
  k_mlp2<4><<<cdiv(G, 16), 32, 0, stream>>>(gfih, G, updW1h, updb1, updW2h, updb2,
                                            (_Float16*)nullptr, gfo);

  // CNN tail
  k_conv1<<<cdiv(64 * 216, 256), 256, 0, stream>>>(gfo, c1w, c1b, act1);
  k_pool1<<<cdiv(64 * 27, 256), 256, 0, stream>>>(act1, y1);
  k_conv2<<<cdiv(128 * 27, 256), 256, 0, stream>>>(y1, c2w, c2b, act2);
  k_pool2<<<1, 128, 0, stream>>>(act2, y2);
  k_conv3<<<1, 256, 0, stream>>>(y2, c3w, c3b, y3);
  k_fc<<<1, 32, 0, stream>>>(y3, fcw, fcb, dout);
}
